// LNMaxPool_41214506172472
// MI455X (gfx1250) — matrix operand order repaired and checked by reading, not compiled, not run on hardware
//
#include <hip/hip_runtime.h>
#include <hip/hip_bf16.h>

// MI455X / gfx1250, wave32. bf16 WMMA path (v_wmma_f32_16x16x32_bf16).

typedef __attribute__((ext_vector_type(16))) __bf16 v16bf;
typedef __attribute__((ext_vector_type(8)))  float  v8f;

#define B_ 16
#define F_ 256
#define K_ 8
#define N_ 1024

__device__ __forceinline__ unsigned short f2bf(float f) {
    unsigned u = __float_as_uint(f);
    u = u + 0x7FFFu + ((u >> 16) & 1u);   // round-to-nearest-even
    return (unsigned short)(u >> 16);
}
__device__ __forceinline__ float bf2f(unsigned short h) {
    return __uint_as_float(((unsigned)h) << 16);
}
// monotonic float -> uint key (total order, works for negatives)
__device__ __forceinline__ unsigned orderKey(float f) {
    unsigned u = __float_as_uint(f);
    return (u & 0x80000000u) ? ~u : (u | 0x80000000u);
}

// ---------------------------------------------------------------- prep:
__global__ void ln_prep(const float* __restrict__ W,
                        unsigned short* __restrict__ Wbf,
                        unsigned long long* __restrict__ best) {
    int g = blockIdx.x * 256 + threadIdx.x;      // 0 .. 65535
    Wbf[g] = f2bf(W[g]);
    if (g < B_ * F_) best[g] = 0ull;
}

// ---------------------------------------------------------------- main:
// one workgroup = (b, 16-wide n tile). 256 threads = 8 waves.
// LDS: xT[k][n'][g] bf16, g contiguous, swizzled: idx = (k*16+n')*256 + (g ^ (n'<<4))
__global__ void __launch_bounds__(256)
ln_main(const float* __restrict__ x,
        const unsigned short* __restrict__ Wbf,
        unsigned long long* __restrict__ best) {
    __shared__ __attribute__((aligned(32))) unsigned short xT[K_ * 16 * F_]; // 64 KB

    const int tid  = threadIdx.x;
    const int b    = blockIdx.y;
    const int n0   = blockIdx.x * 16;
    const int wave = tid >> 5;
    const int lane = tid & 31;
    const int np   = lane & 15;   // column n' within tile
    const int hi   = lane >> 4;   // lane half (0/1)

    // ---- stage x tile into LDS as bf16, transposed to g-contiguous ----
    const float* xb = x + (size_t)b * F_ * K_ * N_;
#pragma unroll
    for (int i = 0; i < 8; ++i) {
        int r = tid + i * 256;            // 0..2047 rows of (f,k)
        int f = r >> 3, k = r & 7;
        const float4* src = (const float4*)(xb + ((size_t)(f * K_ + k) * N_) + n0);
        float4 q0 = src[0], q1 = src[1], q2 = src[2], q3 = src[3];
        float vals[16] = {q0.x, q0.y, q0.z, q0.w, q1.x, q1.y, q1.z, q1.w,
                          q2.x, q2.y, q2.z, q2.w, q3.x, q3.y, q3.z, q3.w};
#pragma unroll
        for (int c = 0; c < 16; ++c)
            xT[(k * 16 + c) * F_ + (f ^ (c << 4))] = f2bf(vals[c]);
    }
    __syncthreads();

    // ---- GEMM: D_k[f, n'] = sum_g W[f,g] * x[b,g,k,n']  via 16x16x32 bf16 WMMA.
    // Split-K over lie coords: 2 passes x 4 accumulators to keep VGPR pressure low.
#pragma unroll 1
    for (int ft = 0; ft < 2; ++ft) {
        const int f0   = (wave * 2 + ft) * 16;
        const int frow = f0 + np;         // A-matrix row for this lane
        const int mb   = hi * 8;

        float kfv[8];                     // per C/D slot: running sum of c_k(x)*d_k
#pragma unroll
        for (int v = 0; v < 8; ++v) kfv[v] = 0.0f;

#pragma unroll 1
        for (int kp = 0; kp < 2; ++kp) {
            v8f acc[4];
#pragma unroll
            for (int kk = 0; kk < 4; ++kk)
#pragma unroll
                for (int e = 0; e < 8; ++e) acc[kk][e] = 0.0f;

#pragma unroll 1
            for (int g0 = 0; g0 < F_; g0 += 32) {
                // A fragment (16x32 bf16): per-lane K = g0+16*hi+{0..15}
                union { v16bf v; uint4 q[2]; } A;
                const unsigned short* wrow = Wbf + frow * F_ + g0 + hi * 16;
                A.q[0] = *(const uint4*)(wrow);
                A.q[1] = *(const uint4*)(wrow + 8);

                const int gb = g0 + hi * 16;    // B per-lane K start (contiguous 16)
                const int bswz = ((gb >> 4) ^ np) << 4;
#pragma unroll
                for (int kk = 0; kk < 4; ++kk) {
                    const int k = kp * 4 + kk;
                    const v16bf* bp = (const v16bf*)&xT[(k * 16 + np) * F_ + bswz];
                    acc[kk] = __builtin_amdgcn_wmma_f32_16x16x32_bf16(
                        false, A.v, false, *bp, (short)0, acc[kk], false, false);
                }
            }

            // partial Killing form: kf = 6 * sum_k c_k(x) * d_k with
            // c = { x6, x7, -2x2, 2x3+x5, 2x4, 2x5+x3, x0, x1 }
#pragma unroll
            for (int v = 0; v < 8; ++v) {
                const int f = f0 + mb + v;
                #define XG(kk) bf2f(xT[((kk) * 16 + np) * F_ + (f ^ (np << 4))])
                if (kp == 0) {
                    float x2 = XG(2), x3 = XG(3), x5 = XG(5), x6 = XG(6), x7 = XG(7);
                    kfv[v] += x6 * acc[0][v] + x7 * acc[1][v]
                            - 2.0f * x2 * acc[2][v]
                            + (2.0f * x3 + x5) * acc[3][v];
                } else {
                    float x0 = XG(0), x1 = XG(1), x3 = XG(3), x4 = XG(4), x5 = XG(5);
                    kfv[v] += 2.0f * x4 * acc[0][v]
                            + (2.0f * x5 + x3) * acc[1][v]
                            + x0 * acc[2][v] + x1 * acc[3][v];
                }
                #undef XG
            }
        }

        // ---- per-row argmax over this 16-n tile (first-max tie rule) ----
#pragma unroll
        for (int v = 0; v < 8; ++v) {
            const int f = f0 + mb + v;
            float kf = 6.0f * kfv[v];
            unsigned long long packed =
                ((unsigned long long)orderKey(kf) << 32) |
                (unsigned long long)(0xFFFFFFFFu - (unsigned)(n0 + np));
#pragma unroll
            for (int s = 1; s < 16; s <<= 1) {
                unsigned long long o = __shfl_xor(packed, s, 32);
                if (o > packed) packed = o;
            }
            if (np == 0)
                atomicMax(&best[b * F_ + f], packed);   // global_atomic_max_u64
        }
    }
}

// ---------------------------------------------------------------- gather:
__global__ void ln_gather(const float* __restrict__ x,
                          const unsigned long long* __restrict__ best,
                          float* __restrict__ out) {
    int t = blockIdx.x * 256 + threadIdx.x;   // 0..4095 = (b,f)
    unsigned long long p = best[t];
    unsigned n = 0xFFFFFFFFu - (unsigned)(p & 0xFFFFFFFFull);
    const float* src = x + (size_t)t * K_ * N_ + n;
#pragma unroll
    for (int k = 0; k < K_; ++k)
        out[t * K_ + k] = src[(size_t)k * N_];
}

extern "C" void kernel_launch(void* const* d_in, const int* in_sizes, int n_in,
                              void* d_out, int out_size, void* d_ws, size_t ws_size,
                              hipStream_t stream) {
    const float* x = (const float*)d_in[0];   // [16,256,8,1024] f32
    const float* W = (const float*)d_in[1];   // [256,256] f32
    float* out = (float*)d_out;               // [16,256,8] f32

    unsigned long long* best = (unsigned long long*)d_ws;                  // 32 KB
    unsigned short* Wbf = (unsigned short*)((char*)d_ws + B_ * F_ * 8);    // 128 KB

    ln_prep<<<(F_ * F_) / 256, 256, 0, stream>>>(W, Wbf, best);
    ln_main<<<dim3(N_ / 16, B_), 256, 0, stream>>>(x, Wbf, best);
    ln_gather<<<(B_ * F_) / 256, 256, 0, stream>>>(x, best, out);
}
